// HetConv_40827959115793
// MI455X (gfx1250) — compile-verified, matrix operand order checked by reading, not run on hardware
//
#include <hip/hip_runtime.h>
#include <math.h>

#define N_NODES   100000
#define N_EDGES   600000
#define HID       128
#define N_ETYPES  10
#define NEG_SLOPE 0.02f
#define BN_EPS    1e-5f

typedef __attribute__((ext_vector_type(16))) _Float16 v16h;
typedef __attribute__((ext_vector_type(8)))  float    v8f;

// ---------------- utility: zero a float buffer ----------------
__global__ void zero_kernel(float* p, int n) {
  int i = blockIdx.x * blockDim.x + threadIdx.x;
  int stride = gridDim.x * blockDim.x;
  for (; i < n; i += stride) p[i] = 0.0f;
}

// ---------------- per-edge-type scalar: ee[t] = sum_k emb*fc*attn ----------------
__global__ void ee_kernel(const float* __restrict__ edge_emb,
                          const float* __restrict__ edges_fc,
                          const float* __restrict__ edges_attn,
                          float* __restrict__ ee) {
  int t = threadIdx.x;
  if (t < N_ETYPES) {
    float s = 0.f;
    for (int k = 0; k < HID; ++k)
      s += edge_emb[t * HID + k] * edges_fc[t * HID + k] * edges_attn[k];
    ee[t] = s;
  }
}

// ---------------- per-node attention scalar: ft[n] = sum_k feat*fc0*attn ----------------
__global__ void ft_kernel(const float* __restrict__ nodes_feat,
                          const float* __restrict__ nodes_fc,   // row 0 used
                          const float* __restrict__ nodes_attn,
                          float* __restrict__ ft) {
  int warp = (blockIdx.x * blockDim.x + threadIdx.x) >> 5;
  int lane = threadIdx.x & 31;
  if (warp >= N_NODES) return;
  float4 x = ((const float4*)(nodes_feat + (size_t)warp * HID))[lane];
  float4 f = ((const float4*)nodes_fc)[lane];
  float4 a = ((const float4*)nodes_attn)[lane];
  float s = x.x * f.x * a.x + x.y * f.y * a.y + x.z * f.z * a.z + x.w * f.w * a.w;
  #pragma unroll
  for (int off = 16; off >= 1; off >>= 1) s += __shfl_xor(s, off, 32);
  if (lane == 0) ft[warp] = s;
}

// ---------------- edge phase: a = leaky(ft[s]+ft[d]+ee[t]); agg[d] += h[s]*a ----------------
__global__ void edge_kernel(const float* __restrict__ nodes_feat,
                            const float* __restrict__ nodes_fc,   // row 0 used
                            const int*   __restrict__ etype,
                            const int*   __restrict__ src,
                            const int*   __restrict__ dst,
                            const float* __restrict__ ft,
                            const float* __restrict__ ee,
                            float* __restrict__ agg) {
  int warp  = (blockIdx.x * blockDim.x + threadIdx.x) >> 5;
  int lane  = threadIdx.x & 31;
  int nwarp = (gridDim.x * blockDim.x) >> 5;
  float4 f = ((const float4*)nodes_fc)[lane];
  for (int e = warp; e < N_EDGES; e += nwarp) {
    int s = src[e], d = dst[e], t = etype[e];
    float sc = ft[s] + ft[d] + ee[t];
    sc = (sc >= 0.f) ? sc : NEG_SLOPE * sc;
    float4 x = ((const float4*)(nodes_feat + (size_t)s * HID))[lane];
    float* ag = agg + (size_t)d * HID + lane * 4;
    atomicAdd(ag + 0, x.x * f.x * sc);
    atomicAdd(ag + 1, x.y * f.y * sc);
    atomicAdd(ag + 2, x.z * f.z * sc);
    atomicAdd(ag + 3, x.w * f.w * sc);
  }
}

// ---------------- WMMA fragment load (f32 memory -> f16 fragment) ----------------
// ISA 7.12.2 16-bit A/B layout, wave32:
//   lane r in [0,15]  : row/col r, K = k0 + {0..7, 16..23}
//   lane r+16         : row/col r, K = k0 + {8..15, 24..31}
__device__ __forceinline__ v16h load_frag(const float* __restrict__ base,
                                          int row0, int k0, int lane) {
  int r = row0 + (lane & 15);
  int h = (lane >> 4) * 8;
  const float* p = base + (size_t)r * HID + k0 + h;
  v16h f;
  #pragma unroll
  for (int i = 0; i < 16; ++i)
    f[i] = (_Float16)p[(i >> 3) * 16 + (i & 7)];
  return f;
}

// ---------------- z = agg @ lin_w^T + lin_b via v_wmma_f32_16x16x32_f16 ----------------
__global__ void gemm_kernel(const float* __restrict__ agg,
                            const float* __restrict__ lin_w,   // [128,128] row-major (j,k)
                            const float* __restrict__ lin_b,
                            float* __restrict__ z) {
  const int NT = N_NODES / 16;                       // 6250 row tiles (exact)
  int wid = (blockIdx.x * blockDim.x + threadIdx.x) >> 5;
  if (wid >= NT) return;                             // wave-uniform: EXEC all-1 inside
  int lane = threadIdx.x & 31;
  int m0 = wid * 16;

  // A fragments: agg rows m0..m0+15, K = 0..127 in four 16x32 chunks
  v16h a0 = load_frag(agg, m0, 0,  lane);
  v16h a1 = load_frag(agg, m0, 32, lane);
  v16h a2 = load_frag(agg, m0, 64, lane);
  v16h a3 = load_frag(agg, m0, 96, lane);

  int col   = lane & 15;
  int rbase = m0 + ((lane >> 4) * 8);

  #pragma unroll
  for (int j = 0; j < 8; ++j) {
    int n0 = j * 16;
    // B[k][n] = lin_w[n][k] -> identical per-lane gather pattern as A
    v16h b0 = load_frag(lin_w, n0, 0,  lane);
    v16h b1 = load_frag(lin_w, n0, 32, lane);
    v16h b2 = load_frag(lin_w, n0, 64, lane);
    v16h b3 = load_frag(lin_w, n0, 96, lane);
    v8f c = {};
    c = __builtin_amdgcn_wmma_f32_16x16x32_f16(false, a0, false, b0, (short)0, c, false, false);
    c = __builtin_amdgcn_wmma_f32_16x16x32_f16(false, a1, false, b1, (short)0, c, false, false);
    c = __builtin_amdgcn_wmma_f32_16x16x32_f16(false, a2, false, b2, (short)0, c, false, false);
    c = __builtin_amdgcn_wmma_f32_16x16x32_f16(false, a3, false, b3, (short)0, c, false, false);
    float bias = lin_b[n0 + col];
    #pragma unroll
    for (int v = 0; v < 8; ++v)
      z[(size_t)(rbase + v) * HID + n0 + col] = c[v] + bias;
  }
}

// ---------------- deterministic per-column batch stats ----------------
__global__ void stats_kernel(const float* __restrict__ z,
                             float* __restrict__ colsum,
                             float* __restrict__ colsumsq) {
  int j = blockIdx.x;          // one column per block
  int tid = threadIdx.x;
  float s = 0.f, q = 0.f;
  for (int r = tid; r < N_NODES; r += 256) {
    float v = z[(size_t)r * HID + j];
    s += v; q += v * v;
  }
  __shared__ float ss[256];
  __shared__ float sq[256];
  ss[tid] = s; sq[tid] = q;
  __syncthreads();
  for (int off = 128; off >= 1; off >>= 1) {
    if (tid < off) { ss[tid] += ss[tid + off]; sq[tid] += sq[tid + off]; }
    __syncthreads();
  }
  if (tid == 0) { colsum[j] = ss[0]; colsumsq[j] = sq[0]; }
}

// ---------------- fold BN into scale/shift ----------------
__global__ void bnfin_kernel(const float* __restrict__ colsum,
                             const float* __restrict__ colsumsq,
                             const float* __restrict__ gamma,
                             const float* __restrict__ beta,
                             float* __restrict__ scale,
                             float* __restrict__ shift) {
  int j = threadIdx.x;
  float mean = colsum[j] * (1.0f / (float)N_NODES);
  float var  = colsumsq[j] * (1.0f / (float)N_NODES) - mean * mean;
  float inv  = rsqrtf(var + BN_EPS);
  float sc   = gamma[j] * inv;
  scale[j] = sc;
  shift[j] = beta[j] - mean * sc;
}

// ---------------- apply BN + ELU in place on d_out ----------------
__global__ void apply_kernel(float* __restrict__ z,
                             const float* __restrict__ scale,
                             const float* __restrict__ shift) {
  int i = blockIdx.x * blockDim.x + threadIdx.x;
  int stride = gridDim.x * blockDim.x;
  const int n4 = N_NODES * HID / 4;
  for (; i < n4; i += stride) {
    int c4 = i & 31;                    // 32 float4 per 128-wide row
    float4 v  = ((float4*)z)[i];
    float4 sc = ((const float4*)scale)[c4];
    float4 sh = ((const float4*)shift)[c4];
    v.x = v.x * sc.x + sh.x;
    v.y = v.y * sc.y + sh.y;
    v.z = v.z * sc.z + sh.z;
    v.w = v.w * sc.w + sh.w;
    v.x = (v.x > 0.f) ? v.x : expm1f(v.x);
    v.y = (v.y > 0.f) ? v.y : expm1f(v.y);
    v.z = (v.z > 0.f) ? v.z : expm1f(v.z);
    v.w = (v.w > 0.f) ? v.w : expm1f(v.w);
    ((float4*)z)[i] = v;
  }
}

extern "C" void kernel_launch(void* const* d_in, const int* in_sizes, int n_in,
                              void* d_out, int out_size, void* d_ws, size_t ws_size,
                              hipStream_t stream) {
  (void)in_sizes; (void)n_in; (void)out_size; (void)ws_size;
  const float* nodes_feat = (const float*)d_in[0];
  const int*   edges_feat = (const int*)  d_in[1];
  const int*   src        = (const int*)  d_in[2];
  const int*   dst        = (const int*)  d_in[3];
  const float* nodes_fc   = (const float*)d_in[4];   // use row 0
  const float* nodes_attn = (const float*)d_in[5];
  const float* edge_emb   = (const float*)d_in[6];
  const float* edges_fc   = (const float*)d_in[7];
  const float* edges_attn = (const float*)d_in[8];
  const float* lin_w      = (const float*)d_in[9];
  const float* lin_b      = (const float*)d_in[10];
  const float* bn_gamma   = (const float*)d_in[11];
  const float* bn_beta    = (const float*)d_in[12];
  float* z = (float*)d_out;

  // workspace layout (floats): [agg N*H][colsum H][colsumsq H][scale H][shift H][ft N][ee 16]
  float* agg      = (float*)d_ws;
  float* colsum   = agg + (size_t)N_NODES * HID;
  float* colsumsq = colsum + HID;
  float* scale    = colsumsq + HID;
  float* shift    = scale + HID;
  float* ft       = shift + HID;
  float* ee       = ft + N_NODES;

  zero_kernel<<<2048, 256, 0, stream>>>(agg, N_NODES * HID);
  ee_kernel<<<1, 32, 0, stream>>>(edge_emb, edges_fc, edges_attn, ee);
  ft_kernel<<<(N_NODES * 32 + 255) / 256, 256, 0, stream>>>(nodes_feat, nodes_fc, nodes_attn, ft);
  edge_kernel<<<(N_EDGES * 32 + 255) / 256, 256, 0, stream>>>(nodes_feat, nodes_fc, edges_feat,
                                                              src, dst, ft, ee, agg);
  gemm_kernel<<<((N_NODES / 16) + 7) / 8, 256, 0, stream>>>(agg, lin_w, lin_b, z);
  stats_kernel<<<HID, 256, 0, stream>>>(z, colsum, colsumsq);
  bnfin_kernel<<<1, HID, 0, stream>>>(colsum, colsumsq, bn_gamma, bn_beta, scale, shift);
  apply_kernel<<<2048, 256, 0, stream>>>(z, scale, shift);
}
